// GCN_33784212750767
// MI455X (gfx1250) — matrix operand order repaired and checked by reading, not compiled, not run on hardware
//
#include <hip/hip_runtime.h>
#include <hip/hip_bf16.h>

// ---------------------------------------------------------------------------
// Problem constants (from reference): B=8, K=128, S=128, H=256, OUT=128
// f: [256->256->256->256->256] relu between; g: [640->256->256->256->128]
// ---------------------------------------------------------------------------

typedef _Float16 h16_t;
typedef __attribute__((ext_vector_type(8)))  _Float16 v8h_t;
typedef __attribute__((ext_vector_type(16))) _Float16 v16h_t;
typedef __attribute__((ext_vector_type(8)))  float    v8f_t;
typedef __attribute__((ext_vector_type(4)))  unsigned int v4u_t;
typedef __attribute__((ext_vector_type(8)))  int v8i_t;
typedef __attribute__((ext_vector_type(4)))  int v4i_t;

union HF16 { v16h_t v; v8h_t h[2]; };

#define LDH 264   // padded row stride (halves): 264*2B = 528B = 132 dwords -> +4 banks/row

// ---- WMMA fragment helpers (CDNA5 16x16x32 f16 layouts, wave32) ------------
// A (16x32, MxK): lane = m + 16*s ; halves [0..7]=k0+8s.., [8..15]=k0+16+8s..
__device__ __forceinline__ v16h_t load_frag_a(const h16_t* base, int ld, int m0, int k0, int lane) {
  int m = lane & 15, s = lane >> 4;
  const h16_t* p = base + (size_t)(m0 + m) * ld + k0 + 8 * s;
  HF16 u;
  u.h[0] = *(const v8h_t*)p;
  u.h[1] = *(const v8h_t*)(p + 16);
  return u.v;
}
// B (32x16, KxN) from W^T stored [N][K]: lane = n + 16*s ; halves [0..7]=k0+8s.., [8..15]=k0+16+8s..
__device__ __forceinline__ v16h_t load_frag_b(const h16_t* base, int ld, int n0, int k0, int lane) {
  int n = lane & 15, s = lane >> 4;
  const h16_t* p = base + (size_t)(n0 + n) * ld + k0 + 8 * s;
  HF16 u;
  u.h[0] = *(const v8h_t*)p;
  u.h[1] = *(const v8h_t*)(p + 16);
  return u.v;
}

__device__ __forceinline__ v8f_t wmma_f16(v16h_t a, v16h_t b, v8f_t c) {
  return __builtin_amdgcn_wmma_f32_16x16x32_f16(false, a, false, b, (short)0, c, false, false);
}

// ---------------------------------------------------------------------------
// Tensor Data Mover: 2D tile global->LDS with hardware row padding.
// D# built per CDNA5 ISA:
//   group0: dw0 = count(1) ; dw1 = lds byte addr ; dw2/dw3 = global addr, type=2
//   group1: data_size=8B, pad fields, tensor_dim0/1, tile_dim0/1, dim0 stride
// (clang-23 toolchain: 6-arg builtin form)
// ---------------------------------------------------------------------------
__device__ __forceinline__ void tdm_load_2d_padded(
    const void* gsrc, unsigned lds_byte_off,
    unsigned row_units8,        // row length in 8-byte units (tile_dim0 == tensor_dim0)
    unsigned rows,              // tile_dim1 == tensor_dim1
    unsigned stride_units8,     // tensor_dim0_stride in 8-byte units
    unsigned pad_interval_code, // 6 -> pad after every 128 DWORDs (512B)
    unsigned pad_amount_code,   // 3 -> insert 4 DWORDs (16B)
    bool pad_en) {
  unsigned long long ga = (unsigned long long)(size_t)gsrc;
  v4u_t g0;
  g0.x = 1u;                                                   // count=1, user mode
  g0.y = lds_byte_off;                                         // lds_addr
  g0.z = (unsigned)ga;                                         // global_addr[31:0]
  g0.w = (unsigned)((ga >> 32) & 0x01FFFFFFu) | (2u << 30);    // addr[56:32] | type=2
  unsigned dw0 = (3u << 16);                                   // data_size = 8 bytes
  if (pad_en)
    dw0 |= (1u << 20) | ((pad_interval_code & 7u) << 22) | ((pad_amount_code & 127u) << 25);
  v8i_t g1;
  g1[0] = (int)dw0;
  g1[1] = (int)((row_units8 & 0xFFFFu) << 16);                           // tensor_dim0[15:0]
  g1[2] = (int)(((row_units8 >> 16) & 0xFFFFu) | ((rows & 0xFFFFu) << 16)); // dim0 hi | dim1 lo
  g1[3] = (int)(((rows >> 16) & 0xFFFFu) | ((row_units8 & 0xFFFFu) << 16)); // dim1 hi | tile_dim0
  g1[4] = (int)(rows & 0xFFFFu);                                         // tile_dim1 (tile_dim2=0)
  g1[5] = (int)stride_units8;                                            // dim0_stride[31:0]
  g1[6] = 0;                                                             // stride hi, dim1_stride lo
  g1[7] = 0;
  v4i_t zz4 = {};
  v8i_t zz8 = {};
  __builtin_amdgcn_tensor_load_to_lds(g0, g1, zz4, zz4, zz8, 0);
}

// ---------------------------------------------------------------------------
// K0: transpose fp32 [Kd][Nd] weight -> f16 [Nd][Kd]
// ---------------------------------------------------------------------------
__global__ void k_transpose(const float* __restrict__ W, h16_t* __restrict__ WT, int Kd, int Nd) {
  int idx = blockIdx.x * blockDim.x + threadIdx.x;
  if (idx < Kd * Nd) {
    int k = idx / Nd, n = idx % Nd;
    WT[(size_t)n * Kd + k] = (h16_t)W[idx];
  }
}

// ---------------------------------------------------------------------------
// K1: f layer 0 split.  A[b,r,p] = x[b,r]·W0[0:128,p]
//                       C[b,r,p] = x[b,r]·W0[128:256,p] + b0[p]
// ---------------------------------------------------------------------------
__global__ __launch_bounds__(256) void k_f_layer0(
    const float* __restrict__ x, const float* __restrict__ fW0,
    const float* __restrict__ fb0,
    h16_t* __restrict__ Ahat, h16_t* __restrict__ Chat) {
  __shared__ float xs[128];
  int row = blockIdx.x;            // b*128 + r, 1024 blocks
  int p = threadIdx.x;             // 0..255
  if (p < 128) xs[p] = x[(size_t)row * 128 + p];
  __syncthreads();
  float a = 0.f, c = fb0[p];
  #pragma unroll 4
  for (int k = 0; k < 128; ++k) {
    float xv = xs[k];
    a += xv * fW0[(size_t)k * 256 + p];
    c += xv * fW0[(size_t)(k + 128) * 256 + p];
  }
  Ahat[(size_t)row * 256 + p] = (h16_t)a;
  Chat[(size_t)row * 256 + p] = (h16_t)c;
}

// ---------------------------------------------------------------------------
// K2: dominant pair kernel. One workgroup per (b,i); rows = j (128).
//     h0 = relu(A[b,j] + C[b,i]) -> 3 chained WMMA GEMMs -> E[b,i] (mask j!=i)
//     Weights staged to LDS by the Tensor Data Mover with hardware padding.
// ---------------------------------------------------------------------------
__global__ __launch_bounds__(256) void k_pair(
    const h16_t* __restrict__ Ahat, const h16_t* __restrict__ Chat,
    const h16_t* __restrict__ fWT,     // 3 x [256][256]
    const float* __restrict__ fb1, const float* __restrict__ fb2,
    const float* __restrict__ fb3,
    float* __restrict__ E) {
  extern __shared__ char smem[];
  h16_t* hbuf0 = (h16_t*)smem;                       // 128*LDH halves
  h16_t* hbuf1 = hbuf0 + 128 * LDH;                  // 128*LDH halves
  h16_t* Wlds  = hbuf1 + 128 * LDH;                  // 256*LDH halves
  float* biasL = (float*)(Wlds + 256 * LDH);         // 256 f32
  float* Eacc  = biasL + 256;                        // 256 f32
  h16_t* Crow  = (h16_t*)(Eacc + 256);               // 256 halves

  const int tid = threadIdx.x;
  const int lane = tid & 31, wave = tid >> 5;
  const int bi = blockIdx.x;                 // 0..1023
  const int b = bi >> 7, i = bi & 127;
  const int m0 = wave * 16;
  const int s = lane >> 4, n = lane & 15;

  if (tid < 256) {
    Eacc[tid] = 0.0f;
    Crow[tid] = Chat[(size_t)bi * 256 + tid];
  }
  __syncthreads();

  // Build h0 = relu(A[b,j,:] + Crow) into hbuf0 (f16)
  const h16_t* Ab = Ahat + (size_t)b * 128 * 256;
  for (int idx = tid; idx < 128 * 32; idx += 256) {  // 8 halves per unit
    int j = idx >> 5, c8 = (idx & 31) * 8;
    v8h_t av = *(const v8h_t*)(Ab + (size_t)j * 256 + c8);
    v8h_t cv = *(const v8h_t*)(Crow + c8);
    v8h_t r;
    #pragma unroll
    for (int t = 0; t < 8; ++t) {
      float f = (float)av[t] + (float)cv[t];
      r[t] = (h16_t)(f > 0.f ? f : 0.f);
    }
    *(v8h_t*)(hbuf0 + (size_t)j * LDH + c8) = r;
  }

  const unsigned wlds_off = (unsigned)(size_t)(void*)Wlds;  // LDS aperture: addr[31:0]

  h16_t* hin = hbuf0;
  h16_t* hout = hbuf1;
  v8f_t z = {};

  for (int l = 0; l < 3; ++l) {
    const h16_t* Wg = fWT + (size_t)l * 256 * 256;
    // TDM: DMA 256x256 f16 weights into LDS, hardware-padding each 512B row
    // by 16B so rows land at the 528B (LDH) conflict-free stride.
    if (tid == 0) {
      tdm_load_2d_padded(Wg, wlds_off, /*row_units8=*/64, /*rows=*/256,
                         /*stride_units8=*/64, /*pad_interval=*/6, /*pad_amount=*/3, true);
      __builtin_amdgcn_s_wait_tensorcnt(0);
    }
    const float* bl = (l == 0) ? fb1 : ((l == 1) ? fb2 : fb3);
    if (tid < 256) biasL[tid] = bl[tid];
    if (l < 2)  // warm next layer's weights toward L2
      __builtin_prefetch(fWT + (size_t)(l + 1) * 256 * 256 + (size_t)tid * 256, 0, 1);
    __syncthreads();

    v8f_t acc[16];
    #pragma unroll
    for (int nt = 0; nt < 16; ++nt) acc[nt] = z;

    #pragma unroll
    for (int kk = 0; kk < 8; ++kk) {
      v16h_t a = load_frag_a(hin, LDH, m0, kk * 32, lane);
      #pragma unroll
      for (int nt = 0; nt < 16; ++nt) {
        v16h_t bf = load_frag_b(Wlds, LDH, nt * 16, kk * 32, lane);
        acc[nt] = wmma_f16(a, bf, acc[nt]);
      }
    }

    if (l < 2) {
      #pragma unroll
      for (int nt = 0; nt < 16; ++nt) {
        float bv = biasL[nt * 16 + n];
        #pragma unroll
        for (int r = 0; r < 8; ++r) {
          float f = acc[nt][r] + bv;
          hout[(size_t)(m0 + r + 8 * s) * LDH + nt * 16 + n] = (h16_t)(f > 0.f ? f : 0.f);
        }
      }
    } else {
      // Final f layer: reduce over j, excluding the diagonal j == i
      int il = i - m0;
      bool inStrip = (il >= 0) && (il < 16) && ((il >> 3) == s);
      int skip_r = il & 7;
      #pragma unroll
      for (int nt = 0; nt < 16; ++nt) {
        float bv = biasL[nt * 16 + n];
        float sum = 0.f;
        #pragma unroll
        for (int r = 0; r < 8; ++r) {
          float f = acc[nt][r] + bv;
          if (!(inStrip && r == skip_r)) sum += f;
        }
        atomicAdd(&Eacc[nt * 16 + n], sum);   // ds_add_f32
      }
    }
    __syncthreads();
    h16_t* t = hin; hin = hout; hout = t;
  }

  if (tid < 256) E[(size_t)bi * 256 + tid] = Eacc[tid];
}

// ---------------------------------------------------------------------------
// K3: Esum[b,p] = sum_i E[b,i,p]
// ---------------------------------------------------------------------------
__global__ __launch_bounds__(256) void k_esum(const float* __restrict__ E, float* __restrict__ Esum) {
  int b = blockIdx.x, p = threadIdx.x;
  float sum = 0.f;
  for (int i = 0; i < 128; ++i) sum += E[((size_t)b * 128 + i) * 256 + p];
  Esum[b * 256 + p] = sum;
}

// ---------------------------------------------------------------------------
// K4: Gin[row] = f16 concat(x[row](128), E[row](256), Esum[b](256))
// ---------------------------------------------------------------------------
__global__ __launch_bounds__(256) void k_build_gin(
    const float* __restrict__ x, const float* __restrict__ E,
    const float* __restrict__ Esum, h16_t* __restrict__ Gin) {
  int row = blockIdx.x, t = threadIdx.x, b = row >> 7;
  h16_t* g = Gin + (size_t)row * 640;
  if (t < 128) g[t] = (h16_t)x[(size_t)row * 128 + t];
  g[128 + t] = (h16_t)E[(size_t)row * 256 + t];
  g[384 + t] = (h16_t)Esum[b * 256 + t];
}

// ---------------------------------------------------------------------------
// K5: g MLP, 8 workgroups (one per batch, 128 rows each), WMMA throughout
// ---------------------------------------------------------------------------
__global__ __launch_bounds__(256) void k_g_mlp(
    const h16_t* __restrict__ Gin,
    const h16_t* __restrict__ gW0T, const h16_t* __restrict__ gW1T,
    const h16_t* __restrict__ gW2T, const h16_t* __restrict__ gW3T,
    const float* __restrict__ gb0, const float* __restrict__ gb1,
    const float* __restrict__ gb2, const float* __restrict__ gb3,
    float* __restrict__ out) {
  extern __shared__ char smem[];
  h16_t* hb0 = (h16_t*)smem;
  h16_t* hb1 = hb0 + 128 * LDH;
  float* biasL = (float*)(hb1 + 128 * LDH);

  const int tid = threadIdx.x, lane = tid & 31, wave = tid >> 5;
  const int m0 = wave * 16;
  const size_t row0 = (size_t)blockIdx.x * 128;
  const int s = lane >> 4, n = lane & 15;
  v8f_t z = {};

  // ---- layer 0: K=640, A from global Gin, B from global gW0T ----
  if (tid < 256) biasL[tid] = gb0[tid];
  __syncthreads();
  {
    v8f_t acc[16];
    #pragma unroll
    for (int nt = 0; nt < 16; ++nt) acc[nt] = z;
    const h16_t* Abase = Gin + row0 * 640;
    for (int kk = 0; kk < 20; ++kk) {
      v16h_t a = load_frag_a(Abase, 640, m0, kk * 32, lane);
      #pragma unroll
      for (int nt = 0; nt < 16; ++nt)
        acc[nt] = wmma_f16(a, load_frag_b(gW0T, 640, nt * 16, kk * 32, lane), acc[nt]);
    }
    #pragma unroll
    for (int nt = 0; nt < 16; ++nt) {
      float bv = biasL[nt * 16 + n];
      #pragma unroll
      for (int r = 0; r < 8; ++r) {
        float f = acc[nt][r] + bv;
        hb0[(size_t)(m0 + r + 8 * s) * LDH + nt * 16 + n] = (h16_t)(f > 0.f ? f : 0.f);
      }
    }
  }
  __syncthreads();

  // ---- layers 1 & 2: K=256, A from LDS, B from global ----
  h16_t* hin = hb0; h16_t* hout = hb1;
  for (int l = 0; l < 2; ++l) {
    const h16_t* Wg = (l == 0) ? gW1T : gW2T;
    const float* bl = (l == 0) ? gb1 : gb2;
    if (tid < 256) biasL[tid] = bl[tid];
    __syncthreads();
    v8f_t acc[16];
    #pragma unroll
    for (int nt = 0; nt < 16; ++nt) acc[nt] = z;
    #pragma unroll
    for (int kk = 0; kk < 8; ++kk) {
      v16h_t a = load_frag_a(hin, LDH, m0, kk * 32, lane);
      #pragma unroll
      for (int nt = 0; nt < 16; ++nt)
        acc[nt] = wmma_f16(a, load_frag_b(Wg, 256, nt * 16, kk * 32, lane), acc[nt]);
    }
    #pragma unroll
    for (int nt = 0; nt < 16; ++nt) {
      float bv = biasL[nt * 16 + n];
      #pragma unroll
      for (int r = 0; r < 8; ++r) {
        float f = acc[nt][r] + bv;
        hout[(size_t)(m0 + r + 8 * s) * LDH + nt * 16 + n] = (h16_t)(f > 0.f ? f : 0.f);
      }
    }
    __syncthreads();
    h16_t* t = hin; hin = hout; hout = t;
  }

  // ---- layer 3: N=128, no relu, f32 straight to out ----
  if (tid < 128) biasL[tid] = gb3[tid];
  __syncthreads();
  {
    v8f_t acc[8];
    #pragma unroll
    for (int nt = 0; nt < 8; ++nt) acc[nt] = z;
    #pragma unroll
    for (int kk = 0; kk < 8; ++kk) {
      v16h_t a = load_frag_a(hin, LDH, m0, kk * 32, lane);
      #pragma unroll
      for (int nt = 0; nt < 8; ++nt)
        acc[nt] = wmma_f16(a, load_frag_b(gW3T, 256, nt * 16, kk * 32, lane), acc[nt]);
    }
    #pragma unroll
    for (int nt = 0; nt < 8; ++nt) {
      float bv = biasL[nt * 16 + n];
      #pragma unroll
      for (int r = 0; r < 8; ++r)
        out[(row0 + m0 + r + 8 * s) * 128 + nt * 16 + n] = acc[nt][r] + bv;
    }
  }
}

// ---------------------------------------------------------------------------
// Workspace layout (bytes)
// ---------------------------------------------------------------------------
static const size_t OFF_FWT  = 0;        // 3*256*256 f16   = 393216
static const size_t OFF_GW0T = 393216;   // 256*640  f16    = 327680
static const size_t OFF_GW1T = 720896;   // 256*256  f16    = 131072
static const size_t OFF_GW2T = 851968;   // 256*256  f16    = 131072
static const size_t OFF_GW3T = 983040;   // 128*256  f16    = 65536
static const size_t OFF_AHAT = 1048576;  // 8*128*256 f16   = 524288
static const size_t OFF_CHAT = 1572864;  // 8*128*256 f16   = 524288
static const size_t OFF_E    = 2097152;  // 8*128*256 f32   = 1048576
static const size_t OFF_ESUM = 3145728;  // 8*256 f32       = 8192
static const size_t OFF_GIN  = 3153920;  // 8*128*640 f16   = 1310720

extern "C" void kernel_launch(void* const* d_in, const int* in_sizes, int n_in,
                              void* d_out, int out_size, void* d_ws, size_t ws_size,
                              hipStream_t stream) {
  (void)in_sizes; (void)n_in; (void)out_size; (void)ws_size;
  const float* x   = (const float*)d_in[0];
  const float* fW0 = (const float*)d_in[1];  const float* fb0 = (const float*)d_in[2];
  const float* fW1 = (const float*)d_in[3];  const float* fb1 = (const float*)d_in[4];
  const float* fW2 = (const float*)d_in[5];  const float* fb2 = (const float*)d_in[6];
  const float* fW3 = (const float*)d_in[7];  const float* fb3 = (const float*)d_in[8];
  const float* gW0 = (const float*)d_in[9];  const float* gb0 = (const float*)d_in[10];
  const float* gW1 = (const float*)d_in[11]; const float* gb1 = (const float*)d_in[12];
  const float* gW2 = (const float*)d_in[13]; const float* gb2 = (const float*)d_in[14];
  const float* gW3 = (const float*)d_in[15]; const float* gb3 = (const float*)d_in[16];
  float* out = (float*)d_out;

  char* ws = (char*)d_ws;
  h16_t* fWT  = (h16_t*)(ws + OFF_FWT);
  h16_t* gW0T = (h16_t*)(ws + OFF_GW0T);
  h16_t* gW1T = (h16_t*)(ws + OFF_GW1T);
  h16_t* gW2T = (h16_t*)(ws + OFF_GW2T);
  h16_t* gW3T = (h16_t*)(ws + OFF_GW3T);
  h16_t* Ahat = (h16_t*)(ws + OFF_AHAT);
  h16_t* Chat = (h16_t*)(ws + OFF_CHAT);
  float* E    = (float*)(ws + OFF_E);
  float* Esum = (float*)(ws + OFF_ESUM);
  h16_t* Gin  = (h16_t*)(ws + OFF_GIN);

  // Weight prep (f16 transposed)
  k_transpose<<<(256 * 256 + 255) / 256, 256, 0, stream>>>(fW1, fWT,                256, 256);
  k_transpose<<<(256 * 256 + 255) / 256, 256, 0, stream>>>(fW2, fWT + 256 * 256,    256, 256);
  k_transpose<<<(256 * 256 + 255) / 256, 256, 0, stream>>>(fW3, fWT + 2 * 256 * 256, 256, 256);
  k_transpose<<<(640 * 256 + 255) / 256, 256, 0, stream>>>(gW0, gW0T, 640, 256);
  k_transpose<<<(256 * 256 + 255) / 256, 256, 0, stream>>>(gW1, gW1T, 256, 256);
  k_transpose<<<(256 * 256 + 255) / 256, 256, 0, stream>>>(gW2, gW2T, 256, 256);
  k_transpose<<<(256 * 128 + 255) / 256, 256, 0, stream>>>(gW3, gW3T, 256, 128);

  // f layer-0 split
  k_f_layer0<<<1024, 256, 0, stream>>>(x, fW0, fb0, Ahat, Chat);

  // Pair kernel: 272,896 B dynamic LDS (fits 320 KB WGP budget)
  size_t pair_lds = (size_t)(2 * 128 * LDH + 256 * LDH) * sizeof(h16_t)
                  + 512 * sizeof(float) + 256 * sizeof(h16_t);
  k_pair<<<1024, 256, pair_lds, stream>>>(Ahat, Chat, fWT, fb1, fb2, fb3, E);

  k_esum<<<8, 256, 0, stream>>>(E, Esum);
  k_build_gin<<<1024, 256, 0, stream>>>(x, E, Esum, Gin);

  size_t g_lds = (size_t)(2 * 128 * LDH) * sizeof(h16_t) + 256 * sizeof(float);
  k_g_mlp<<<8, 256, g_lds, stream>>>(Gin, gW0T, gW1T, gW2T, gW3T,
                                     gb0, gb1, gb2, gb3, out);
}
